// CRANet_29360396435858
// MI455X (gfx1250) — matrix operand order repaired and checked by reading, not compiled, not run on hardware
//
#include <hip/hip_runtime.h>
#include <hip/hip_bf16.h>
#include <math.h>

// ---------------------------------------------------------------------------
// CDNA5 (gfx1250) WMMA types & helpers
// ---------------------------------------------------------------------------
typedef __attribute__((ext_vector_type(16))) _Float16 v16h;
typedef __attribute__((ext_vector_type(8)))  _Float16 v8h;
typedef __attribute__((ext_vector_type(8)))  float    v8f;

#define DEVFN static __device__ __forceinline__

DEVFN float eluf(float x) { return x > 0.f ? x : (__expf(x) - 1.f); }
DEVFN float sigf(float x) { return 1.f / (1.f + __expf(-x)); }
DEVFN v8f vzero() { v8f z = {0.f,0.f,0.f,0.f,0.f,0.f,0.f,0.f}; return z; }

DEVFN v8f wmma16(v16h a, v16h b, v8f c) {
  // emits v_wmma_f32_16x16x32_f16 (probe-confirmed 8-arg signature)
  return __builtin_amdgcn_wmma_f32_16x16x32_f16(false, a, false, b, (short)0, c, false, false);
}

// A fragment (16x32 f16, M x K) from an LDS tile, row-major, row stride in halves.
// ISA layout: lanes 0-15 m=0..15 elems[0..7]=K 0..7, elems[8..15]=K 16..23;
//             lanes 16-31 same m, K shifted by +8.
DEVFN v16h load_a(const _Float16* tile, int stride, int kbase, int lane) {
  int m = lane & 15, hi = lane >> 4;
  const _Float16* row = tile + m * stride + kbase + 8 * hi;
  union { v16h v; v8h g[2]; } t;
  t.g[0] = *(const v8h*)(row);
  t.g[1] = *(const v8h*)(row + 16);
  return t.v;
}
DEVFN v16h load_a_scaled(const _Float16* tile, int stride, int kbase, int lane, float s) {
  v16h a = load_a(tile, stride, kbase, lane);
#pragma unroll
  for (int i = 0; i < 16; ++i) a[i] = (_Float16)((float)a[i] * s);
  return a;
}
// B fragment (32x16 f16, K x N) from transposed weights stored [Npad][Kpad] in global.
// ISA layout: lanes 0-15 n=lane, elems[0..7]=K 0..7, elems[8..15]=K 16..23; lanes 16-31 n=lane-16, K shifted by +8.
DEVFN v16h load_b(const _Float16* wt, int Kpad, int nt, int kbase, int lane) {
  int lo = lane & 15, hi = lane >> 4;
  const _Float16* p = wt + (size_t)(nt * 16 + lo) * (size_t)Kpad + kbase + 8 * hi;
  union { v16h v; v8h g[2]; } t;
  t.g[0] = *(const v8h*)(p);
  t.g[1] = *(const v8h*)(p + 16);
  return t.v;
}
// Store C fragment (+bias, optional elu) as f16 into an LDS tile (row stride halves).
// C layout: elem f -> row m=f+8*hi, col n=nt*16+(lane&15).
DEVFN void store_c(v8f c, const float* bias, int nb, int nt,
                   _Float16* out, int stride, int lane, bool elu_act) {
  int lo = lane & 15, hi = lane >> 4;
  int n = nt * 16 + lo;
  float b = (n < nb) ? bias[n] : 0.f;
#pragma unroll
  for (int f = 0; f < 8; ++f) {
    float x = c[f] + b;
    if (elu_act) x = eluf(x);
    out[(f + 8 * hi) * stride + n] = (_Float16)x;
  }
}

// ---------------------------------------------------------------------------
// Problem constants
// ---------------------------------------------------------------------------
static const int Rr = 1024, Ss = 64, Vv = 8;
static const int RS = Rr * Ss;          // 65536
// P = RS * Vv = 524288 points

// ---------------------------------------------------------------------------
// Kernel 0: transpose+pad weights to f16 [Npad][Kpad]
// ---------------------------------------------------------------------------
__global__ void k_wcvt(const float* __restrict__ W, _Float16* __restrict__ Wt,
                       int K, int N, int Kpad, int Npad) {
  int total = Kpad * Npad;
  for (int i = blockIdx.x * blockDim.x + threadIdx.x; i < total; i += gridDim.x * blockDim.x) {
    int n = i / Kpad, k = i - n * Kpad;
    float v = (k < K && n < N) ? W[(size_t)k * N + n] : 0.f;
    Wt[i] = (_Float16)v;
  }
}

// ---------------------------------------------------------------------------
// Kernel 1: per-(r,s) prelude: ray-diff MLP, neuray gate, weighted mean/var over V
// ---------------------------------------------------------------------------
__global__ void k_prep(const float* __restrict__ rgb_feat, const float* __restrict__ neuray,
                       const float* __restrict__ ray_diff, const float* __restrict__ mask,
                       const float* rd_w1, const float* rd_b1, const float* rd_w2, const float* rd_b2,
                       const float* nr_w1, const float* nr_b1, const float* nr_w2, const float* nr_b2,
                       _Float16* __restrict__ g_buf, _Float16* __restrict__ feat_buf,
                       float* __restrict__ w_buf, float* __restrict__ summask_buf) {
  __shared__ float s_rdw1[64], s_rdb1[16], s_rdw2[560], s_rdb2[35];
  __shared__ float s_nrw1[256], s_nrb1[8], s_nrw2[8], s_nrb2[1];
  int tid = threadIdx.x;
  for (int i = tid; i < 64;  i += blockDim.x) s_rdw1[i] = rd_w1[i];
  for (int i = tid; i < 16;  i += blockDim.x) s_rdb1[i] = rd_b1[i];
  for (int i = tid; i < 560; i += blockDim.x) s_rdw2[i] = rd_w2[i];
  for (int i = tid; i < 35;  i += blockDim.x) s_rdb2[i] = rd_b2[i];
  for (int i = tid; i < 256; i += blockDim.x) s_nrw1[i] = nr_w1[i];
  for (int i = tid; i < 8;   i += blockDim.x) s_nrb1[i] = nr_b1[i];
  for (int i = tid; i < 8;   i += blockDim.x) s_nrw2[i] = nr_w2[i];
  if (tid == 0) s_nrb2[0] = nr_b2[0];
  __syncthreads();

  int rs = blockIdx.x * blockDim.x + tid;
  size_t base = (size_t)rs * Vv;

  float mk[8], summ = 0.f;
  for (int v = 0; v < 8; ++v) { mk[v] = mask[base + v]; summ += mk[v]; }
  float winv = 1.f / (summ + 1e-8f);
  float w[8], w0[8];
  for (int v = 0; v < 8; ++v) w[v] = mk[v] * winv;

  for (int v = 0; v < 8; ++v) {
    const float* nf = neuray + (base + v) * 32;
    float h[8];
    for (int o = 0; o < 8; ++o) {
      float a = s_nrb1[o];
      for (int i = 0; i < 32; ++i) a += nf[i] * s_nrw1[i * 8 + o];
      h[o] = eluf(a);
    }
    float s = s_nrb2[0];
    for (int i = 0; i < 8; ++i) s += h[i] * s_nrw2[i];
    w0[v] = sigf(s) * w[v];
  }

  float m0[35], m1[35];
  for (int j = 0; j < 35; ++j) { m0[j] = 0.f; m1[j] = 0.f; }
  for (int v = 0; v < 8; ++v) {
    const float* rd = ray_diff + (base + v) * 4;
    float hh[16];
    for (int o = 0; o < 16; ++o) {
      float a = s_rdb1[o];
      for (int i = 0; i < 4; ++i) a += rd[i] * s_rdw1[i * 16 + o];
      hh[o] = eluf(a);
    }
    const float* rf = rgb_feat + (base + v) * 35;
    _Float16* fb = feat_buf + (base + v) * 36;
    for (int j = 0; j < 35; ++j) {
      float a = s_rdb2[j];
      for (int i = 0; i < 16; ++i) a += hh[i] * s_rdw2[i * 35 + j];
      float f = rf[j] + eluf(a);
      fb[j] = (_Float16)f;
      m0[j] += w0[v] * f;
      m1[j] += w[v] * f;
    }
    fb[35] = (_Float16)0.f;
  }

  float v0[35], v1[35];
  for (int j = 0; j < 35; ++j) { v0[j] = 0.f; v1[j] = 0.f; }
  for (int v = 0; v < 8; ++v) {
    const _Float16* fb = feat_buf + (base + v) * 36;
    for (int j = 0; j < 35; ++j) {
      float f = (float)fb[j];
      float d0 = f - m0[j]; v0[j] += w0[v] * d0 * d0;
      float d1 = f - m1[j]; v1[j] += w[v] * d1 * d1;
    }
  }

  _Float16* g = g_buf + (size_t)rs * 144;
  for (int j = 0; j < 35; ++j) {
    g[j]       = (_Float16)m0[j];
    g[35 + j]  = (_Float16)v0[j];
    g[70 + j]  = (_Float16)m1[j];
    g[105 + j] = (_Float16)v1[j];
  }
  for (int j = 140; j < 144; ++j) g[j] = (_Float16)0.f;
  for (int v = 0; v < 8; ++v) w_buf[base + v] = w[v];
  summask_buf[rs] = summ;
}

// ---------------------------------------------------------------------------
// Kernel 2: WMMA MLP chain. 1 wave = 16 points; 4 waves/block.
// ---------------------------------------------------------------------------
struct MlpArgs {
  const float* neuray;
  const float* mask;
  const float* ray_diff;
  const float *base_b1, *base_b2, *vis_b1, *vis_b2, *v2_b1, *v2_b2, *rgb_b1, *rgb_b2, *rgb_b3;
  const _Float16 *W1t, *W2t, *VW1t, *VW2t, *U1t, *U2t, *R1t, *R2t, *R3t;
  const _Float16 *g_buf, *feat_buf;
  const float* w_buf;
  float *x_buf, *visf_buf, *blend_buf;
};

__global__ void __launch_bounds__(128) k_mlp(MlpArgs A) {
  __shared__ _Float16 sBig[4][16 * 224];  // input tile, later carved into X/T/U (stride 64)
  __shared__ _Float16 sH[4][16 * 64];     // hidden tile
  __shared__ float sVm[4][16];            // per-row vis_mid
  __shared__ float sVf[4][16];            // per-row vis_final

  int lane = threadIdx.x & 31;
  int wvi  = threadIdx.x >> 5;
  int tile = blockIdx.x * 4 + wvi;
  size_t P0 = (size_t)tile * 16;
  int m = lane & 15, hi = lane >> 4;
  size_t pm = P0 + (size_t)m;

  _Float16* BIG = sBig[wvi];
  _Float16* X = BIG;            // 16x64 (cols 0..31 live)
  _Float16* T = BIG + 1024;     // 16x64 scratch
  _Float16* U = BIG + 2048;     // 16x64 scratch
  _Float16* H = sH[wvi];        // 16x64

  // ---- stage input [g(140) | feat(35) | neuray(32) | pad] -> BIG (stride 224)
  {
    size_t rs = pm >> 3;
    const _Float16* g  = A.g_buf   + rs * 144;
    const _Float16* ft = A.feat_buf + pm * 36;
    const float*    nf = A.neuray  + pm * 32;
    _Float16* dst = BIG + m * 224;
    int c0 = hi * 112;
    for (int c = c0; c < c0 + 112; ++c) {
      _Float16 val;
      if      (c < 140) val = g[c];
      else if (c < 175) val = ft[c - 140];
      else if (c < 207) val = (_Float16)nf[c - 175];
      else              val = (_Float16)0.f;
      dst[c] = val;
    }
  }
  __syncthreads();

  // ---- base layer1: 224(207) -> 64, elu -> H
  {
    v8f acc[4] = {vzero(), vzero(), vzero(), vzero()};
    for (int kt = 0; kt < 7; ++kt) {
      v16h a = load_a(BIG, 224, kt * 32, lane);
#pragma unroll
      for (int nt = 0; nt < 4; ++nt)
        acc[nt] = wmma16(a, load_b(A.W1t, 224, nt, kt * 32, lane), acc[nt]);
    }
#pragma unroll
    for (int nt = 0; nt < 4; ++nt) store_c(acc[nt], A.base_b1, 64, nt, H, 64, lane, true);
  }
  __syncthreads();

  // ---- base layer2: 64 -> 32, elu -> X
  {
    v8f acc[2] = {vzero(), vzero()};
#pragma unroll
    for (int kt = 0; kt < 2; ++kt) {
      v16h a = load_a(H, 64, kt * 32, lane);
#pragma unroll
      for (int nt = 0; nt < 2; ++nt)
        acc[nt] = wmma16(a, load_b(A.W2t, 64, nt, kt * 32, lane), acc[nt]);
    }
#pragma unroll
    for (int nt = 0; nt < 2; ++nt) store_c(acc[nt], A.base_b2, 32, nt, X, 64, lane, true);
  }
  __syncthreads();

  float w_m  = A.w_buf[pm];
  float mk_m = A.mask[pm];

  // ---- vis layer1: (x*w) 32 -> 32, elu -> T
  {
    v16h a = load_a_scaled(X, 64, 0, lane, w_m);
    v8f acc[2] = {vzero(), vzero()};
#pragma unroll
    for (int nt = 0; nt < 2; ++nt)
      acc[nt] = wmma16(a, load_b(A.VW1t, 32, nt, 0, lane), acc[nt]);
#pragma unroll
    for (int nt = 0; nt < 2; ++nt) store_c(acc[nt], A.vis_b1, 32, nt, T, 64, lane, true);
  }
  __syncthreads();

  // ---- vis layer2: 32 -> 48(33), elu -> U
  {
    v16h a = load_a(T, 64, 0, lane);
    v8f acc[3] = {vzero(), vzero(), vzero()};
#pragma unroll
    for (int nt = 0; nt < 3; ++nt)
      acc[nt] = wmma16(a, load_b(A.VW2t, 32, nt, 0, lane), acc[nt]);
#pragma unroll
    for (int nt = 0; nt < 3; ++nt) store_c(acc[nt], A.vis_b2, 33, nt, U, 64, lane, true);
  }
  __syncthreads();

  // ---- residual update x += x_res; vis_mid = sig(col32)*mask; emit x
  {
    if (hi == 0) sVm[wvi][m] = sigf((float)U[m * 64 + 32]) * mk_m;
    for (int c = 0; c < 16; ++c) {
      int idx = m * 64 + hi * 16 + c;
      float xv = (float)X[idx] + (float)U[idx];
      X[idx] = (_Float16)xv;
      A.x_buf[pm * 32 + hi * 16 + c] = xv;
    }
  }
  __syncthreads();

  // ---- v2 hidden: (x*vis_mid) 32 -> 32, elu -> T
  {
    v16h a = load_a_scaled(X, 64, 0, lane, sVm[wvi][m]);
    v8f acc[2] = {vzero(), vzero()};
#pragma unroll
    for (int nt = 0; nt < 2; ++nt)
      acc[nt] = wmma16(a, load_b(A.U1t, 32, nt, 0, lane), acc[nt]);
#pragma unroll
    for (int nt = 0; nt < 2; ++nt) store_c(acc[nt], A.v2_b1, 32, nt, T, 64, lane, true);
  }
  __syncthreads();

  // ---- v2 out: 32 -> 1 (col 0) -> U
  {
    v16h a = load_a(T, 64, 0, lane);
    v8f acc = vzero();
    acc = wmma16(a, load_b(A.U2t, 32, 0, 0, lane), acc);
    store_c(acc, A.v2_b2, 1, 0, U, 64, lane, false);
  }
  __syncthreads();

  if (hi == 0) {
    float vf = sigf((float)U[m * 64]) * mk_m;
    sVf[wvi][m] = vf;
    A.visf_buf[pm] = vf;
  }
  __syncthreads();

  // ---- stage xr = [x(32), visf, ray_diff(4), 0...] -> T
  {
    const float* rd = A.ray_diff + pm * 4;
    float vf = sVf[wvi][m];
    int c0 = hi * 32;
    for (int c = c0; c < c0 + 32; ++c) {
      _Float16 val;
      if      (c < 32)  val = X[m * 64 + c];
      else if (c == 32) val = (_Float16)vf;
      else if (c < 37)  val = (_Float16)rd[c - 33];
      else              val = (_Float16)0.f;
      T[m * 64 + c] = val;
    }
  }
  __syncthreads();

  // ---- rgb layer1: 64(37) -> 16, elu -> U (zero cols 16..31 for next K-pad)
  {
    v8f acc = vzero();
#pragma unroll
    for (int kt = 0; kt < 2; ++kt)
      acc = wmma16(load_a(T, 64, kt * 32, lane), load_b(A.R1t, 64, 0, kt * 32, lane), acc);
    store_c(acc, A.rgb_b1, 16, 0, U, 64, lane, true);
    for (int c = 0; c < 8; ++c) U[m * 64 + 16 + hi * 8 + c] = (_Float16)0.f;
  }
  __syncthreads();

  // ---- rgb layer2: 32(16) -> 8, elu -> T (zero cols 16..31)
  {
    v8f acc = vzero();
    acc = wmma16(load_a(U, 64, 0, lane), load_b(A.R2t, 32, 0, 0, lane), acc);
    store_c(acc, A.rgb_b2, 8, 0, T, 64, lane, true);
    for (int c = 0; c < 8; ++c) T[m * 64 + 16 + hi * 8 + c] = (_Float16)0.f;
  }
  __syncthreads();

  // ---- rgb layer3: 32(8) -> 1 -> U, emit blend logit
  {
    v8f acc = vzero();
    acc = wmma16(load_a(T, 64, 0, lane), load_b(A.R3t, 32, 0, 0, lane), acc);
    store_c(acc, A.rgb_b3, 1, 0, U, 64, lane, false);
  }
  __syncthreads();
  if (hi == 0) A.blend_buf[pm] = (float)U[m * 64];
}

// ---------------------------------------------------------------------------
// Kernel 3: geo reduction + geo MLP + positional encoding
// ---------------------------------------------------------------------------
__global__ void k_geo(const float* __restrict__ x_buf, const float* __restrict__ visf,
                      const float* gw1, const float* gb1, const float* gw2, const float* gb2,
                      const float* __restrict__ pos_enc, float* __restrict__ gf_buf) {
  __shared__ float s_w1[65 * 64];
  __shared__ float s_b1[64];
  __shared__ float s_w2[64 * 16];
  __shared__ float s_b2[16];
  int tid = threadIdx.x;
  for (int i = tid; i < 65 * 64; i += blockDim.x) s_w1[i] = gw1[i];
  for (int i = tid; i < 64;      i += blockDim.x) s_b1[i] = gb1[i];
  for (int i = tid; i < 64 * 16; i += blockDim.x) s_w2[i] = gw2[i];
  for (int i = tid; i < 16;      i += blockDim.x) s_b2[i] = gb2[i];
  __syncthreads();

  int rs = blockIdx.x * blockDim.x + tid;
  size_t base = (size_t)rs * Vv;

  float vv[8], sv = 0.f;
  for (int v = 0; v < 8; ++v) { vv[v] = visf[base + v]; sv += vv[v]; }
  float inv = 1.f / (sv + 1e-8f), w2s = 0.f;
  for (int v = 0; v < 8; ++v) { vv[v] *= inv; w2s += vv[v]; }

  float m[32];
  for (int j = 0; j < 32; ++j) m[j] = 0.f;
  for (int v = 0; v < 8; ++v) {
    const float* xp = x_buf + (base + v) * 32;
    for (int j = 0; j < 32; ++j) m[j] += xp[j] * vv[v];
  }
  float va[32];
  for (int j = 0; j < 32; ++j) va[j] = 0.f;
  for (int v = 0; v < 8; ++v) {
    const float* xp = x_buf + (base + v) * 32;
    for (int j = 0; j < 32; ++j) { float d = xp[j] - m[j]; va[j] += vv[v] * d * d; }
  }

  float gin[65];
  for (int j = 0; j < 32; ++j) { gin[j] = m[j]; gin[32 + j] = va[j]; }
  gin[64] = w2s * 0.125f;

  float h[64];
  for (int o = 0; o < 64; ++o) {
    float a = s_b1[o];
    for (int i = 0; i < 65; ++i) a += gin[i] * s_w1[i * 64 + o];
    h[o] = eluf(a);
  }
  int s = rs & (Ss - 1);
  float* out = gf_buf + (size_t)rs * 16;
  for (int o = 0; o < 16; ++o) {
    float a = s_b2[o];
    for (int i = 0; i < 64; ++i) a += h[i] * s_w2[i * 16 + o];
    out[o] = eluf(a) + pos_enc[s * 16 + o];
  }
}

// ---------------------------------------------------------------------------
// Kernel 4: per-ray attention (S=64 tokens, 4 heads x d_k 4) + LN + og -> sigma
// ---------------------------------------------------------------------------
__global__ void k_attn(const float* __restrict__ gf_buf, const float* __restrict__ summask,
                       const float* wq, const float* wk, const float* wv, const float* fc,
                       const float* lng, const float* lnb,
                       const float* ogw1, const float* ogb1, const float* ogw2, const float* ogb2,
                       float* __restrict__ sigma_buf) {
  __shared__ float s_wq[256], s_wk[256], s_wv[256], s_fc[256];
  __shared__ float s_lng[16], s_lnb[16], s_ogw1[256], s_ogb1[16], s_ogw2[16], s_ogb2[1];
  __shared__ float sGF[64][16], sK[64][16], sV[64][16], sNV[64];
  int t = threadIdx.x;
  for (int i = t; i < 256; i += 64) {
    s_wq[i] = wq[i]; s_wk[i] = wk[i]; s_wv[i] = wv[i]; s_fc[i] = fc[i]; s_ogw1[i] = ogw1[i];
  }
  if (t < 16) { s_lng[t] = lng[t]; s_lnb[t] = lnb[t]; s_ogb1[t] = ogb1[t]; s_ogw2[t] = ogw2[t]; }
  if (t == 0) s_ogb2[0] = ogb2[0];

  int rs = blockIdx.x * 64 + t;
  const float* g = gf_buf + (size_t)rs * 16;
  for (int i = 0; i < 16; ++i) sGF[t][i] = g[i];
  sNV[t] = summask[rs];
  __syncthreads();

  float q[16];
  for (int o = 0; o < 16; ++o) {
    float aq = 0.f, ak = 0.f, av = 0.f;
    for (int i = 0; i < 16; ++i) {
      float gi = sGF[t][i];
      aq += gi * s_wq[i * 16 + o];
      ak += gi * s_wk[i * 16 + o];
      av += gi * s_wv[i * 16 + o];
    }
    q[o] = aq; sK[t][o] = ak; sV[t][o] = av;
  }
  __syncthreads();

  bool qvalid = sNV[t] > 1.f;
  float o16[16];
  for (int i = 0; i < 16; ++i) o16[i] = 0.f;
  for (int h = 0; h < 4; ++h) {
    float sc[64], mx = -3.4e38f;
    for (int kk = 0; kk < 64; ++kk) {
      float s;
      if (!qvalid) s = -1e9f;
      else {
        s = 0.f;
        for (int d = 0; d < 4; ++d) s += 0.5f * q[h * 4 + d] * sK[kk][h * 4 + d];
      }
      sc[kk] = s; mx = fmaxf(mx, s);
    }
    float den = 0.f;
    for (int kk = 0; kk < 64; ++kk) { float e = __expf(sc[kk] - mx); sc[kk] = e; den += e; }
    float invd = 1.f / den;
    for (int kk = 0; kk < 64; ++kk) {
      float p = sc[kk] * invd;
      for (int d = 0; d < 4; ++d) o16[h * 4 + d] += p * sV[kk][h * 4 + d];
    }
  }

  float oo[16];
  for (int o = 0; o < 16; ++o) {
    float a = 0.f;
    for (int i = 0; i < 16; ++i) a += o16[i] * s_fc[i * 16 + o];
    oo[o] = a + sGF[t][o];
  }
  float mu = 0.f;
  for (int o = 0; o < 16; ++o) mu += oo[o];
  mu *= (1.f / 16.f);
  float va = 0.f;
  for (int o = 0; o < 16; ++o) { float d = oo[o] - mu; va += d * d; }
  va *= (1.f / 16.f);
  float rstd = 1.f / sqrtf(va + 1e-6f);
  float ln[16];
  for (int o = 0; o < 16; ++o) {
    float x = (oo[o] - mu) * rstd * s_lng[o] + s_lnb[o];
    if (x != x) x = 0.f;   // nan_to_num
    ln[o] = x;
  }
  float hh[16];
  for (int o = 0; o < 16; ++o) {
    float a = s_ogb1[o];
    for (int i = 0; i < 16; ++i) a += ln[i] * s_ogw1[i * 16 + o];
    hh[o] = eluf(a);
  }
  float s = s_ogb2[0];
  for (int i = 0; i < 16; ++i) s += hh[i] * s_ogw2[i];
  float sg = fmaxf(s, 0.f);
  if (sNV[t] < 1.f) sg = 0.f;
  sigma_buf[rs] = sg;
}

// ---------------------------------------------------------------------------
// Kernel 5: masked softmax blend over V, pack (R,S,4) output
// ---------------------------------------------------------------------------
__global__ void k_final(const float* __restrict__ rgb_feat, const float* __restrict__ mask,
                        const float* __restrict__ blend, const float* __restrict__ sigma,
                        float* __restrict__ out) {
  int rs = blockIdx.x * blockDim.x + threadIdx.x;
  size_t base = (size_t)rs * Vv;
  float lg[8], mx = -3.4e38f;
  for (int v = 0; v < 8; ++v) {
    float l = blend[base + v];
    if (mask[base + v] == 0.f) l = -1e9f;
    lg[v] = l; mx = fmaxf(mx, l);
  }
  float den = 0.f;
  for (int v = 0; v < 8; ++v) { float e = __expf(lg[v] - mx); lg[v] = e; den += e; }
  float inv = 1.f / den;
  float r = 0.f, g = 0.f, b = 0.f;
  for (int v = 0; v < 8; ++v) {
    float bw = lg[v] * inv;
    const float* rf = rgb_feat + (base + v) * 35;
    r += rf[0] * bw; g += rf[1] * bw; b += rf[2] * bw;
  }
  float* o = out + (size_t)rs * 4;
  o[0] = r; o[1] = g; o[2] = b; o[3] = sigma[rs];
}

// ---------------------------------------------------------------------------
// Launch
// ---------------------------------------------------------------------------
extern "C" void kernel_launch(void* const* d_in, const int* in_sizes, int n_in,
                              void* d_out, int out_size, void* d_ws, size_t ws_size,
                              hipStream_t stream) {
  (void)in_sizes; (void)n_in; (void)out_size; (void)ws_size;
  const float* rgb_feat = (const float*)d_in[0];
  const float* neuray   = (const float*)d_in[1];
  const float* ray_diff = (const float*)d_in[2];
  const float* mask     = (const float*)d_in[3];
  const float* rd_w1 = (const float*)d_in[4];  const float* rd_b1 = (const float*)d_in[5];
  const float* rd_w2 = (const float*)d_in[6];  const float* rd_b2 = (const float*)d_in[7];
  const float* nr_w1 = (const float*)d_in[8];  const float* nr_b1 = (const float*)d_in[9];
  const float* nr_w2 = (const float*)d_in[10]; const float* nr_b2 = (const float*)d_in[11];
  const float* base_w1 = (const float*)d_in[12]; const float* base_b1 = (const float*)d_in[13];
  const float* base_w2 = (const float*)d_in[14]; const float* base_b2 = (const float*)d_in[15];
  const float* vis_w1 = (const float*)d_in[16];  const float* vis_b1 = (const float*)d_in[17];
  const float* vis_w2 = (const float*)d_in[18];  const float* vis_b2 = (const float*)d_in[19];
  const float* v2_w1 = (const float*)d_in[20];   const float* v2_b1 = (const float*)d_in[21];
  const float* v2_w2 = (const float*)d_in[22];   const float* v2_b2 = (const float*)d_in[23];
  const float* geo_w1 = (const float*)d_in[24];  const float* geo_b1 = (const float*)d_in[25];
  const float* geo_w2 = (const float*)d_in[26];  const float* geo_b2 = (const float*)d_in[27];
  const float* wq = (const float*)d_in[28]; const float* wk = (const float*)d_in[29];
  const float* wv = (const float*)d_in[30]; const float* att_fc = (const float*)d_in[31];
  const float* ln_g = (const float*)d_in[32]; const float* ln_b = (const float*)d_in[33];
  const float* og_w1 = (const float*)d_in[34]; const float* og_b1 = (const float*)d_in[35];
  const float* og_w2 = (const float*)d_in[36]; const float* og_b2 = (const float*)d_in[37];
  const float* rgb_w1 = (const float*)d_in[38]; const float* rgb_b1 = (const float*)d_in[39];
  const float* rgb_w2 = (const float*)d_in[40]; const float* rgb_b2 = (const float*)d_in[41];
  const float* rgb_w3 = (const float*)d_in[42]; const float* rgb_b3 = (const float*)d_in[43];
  const float* pos_enc = (const float*)d_in[44];

  const size_t P = (size_t)RS * Vv;
  char* ws = (char*)d_ws;
  size_t off = 0;
  auto alloc = [&](size_t bytes) { size_t o = off; off = (off + bytes + 255) & ~(size_t)255; return o; };

  _Float16* W1t  = (_Float16*)(ws + alloc((size_t)64 * 224 * 2));
  _Float16* W2t  = (_Float16*)(ws + alloc((size_t)32 * 64 * 2));
  _Float16* VW1t = (_Float16*)(ws + alloc((size_t)32 * 32 * 2));
  _Float16* VW2t = (_Float16*)(ws + alloc((size_t)48 * 32 * 2));
  _Float16* U1t  = (_Float16*)(ws + alloc((size_t)32 * 32 * 2));
  _Float16* U2t  = (_Float16*)(ws + alloc((size_t)16 * 32 * 2));
  _Float16* R1t  = (_Float16*)(ws + alloc((size_t)16 * 64 * 2));
  _Float16* R2t  = (_Float16*)(ws + alloc((size_t)16 * 32 * 2));
  _Float16* R3t  = (_Float16*)(ws + alloc((size_t)16 * 32 * 2));
  _Float16* g_buf    = (_Float16*)(ws + alloc((size_t)RS * 144 * 2));
  _Float16* feat_buf = (_Float16*)(ws + alloc(P * 36 * 2));
  float* w_buf       = (float*)(ws + alloc(P * 4));
  float* summask_buf = (float*)(ws + alloc((size_t)RS * 4));
  float* x_buf       = (float*)(ws + alloc(P * 32 * 4));
  float* visf_buf    = (float*)(ws + alloc(P * 4));
  float* blend_buf   = (float*)(ws + alloc(P * 4));
  float* gf_buf      = (float*)(ws + alloc((size_t)RS * 16 * 4));
  float* sigma_buf   = (float*)(ws + alloc((size_t)RS * 4));

  // Weight transposition to f16 [Npad][Kpad]
  auto cvt = [&](const float* W, _Float16* Wt, int K, int N, int Kpad, int Npad) {
    int total = Kpad * Npad;
    int grid = (total + 255) / 256;
    k_wcvt<<<grid, 256, 0, stream>>>(W, Wt, K, N, Kpad, Npad);
  };
  cvt(base_w1, W1t, 207, 64, 224, 64);
  cvt(base_w2, W2t, 64, 32, 64, 32);
  cvt(vis_w1, VW1t, 32, 32, 32, 32);
  cvt(vis_w2, VW2t, 32, 33, 32, 48);
  cvt(v2_w1,  U1t, 32, 32, 32, 32);
  cvt(v2_w2,  U2t, 32, 1, 32, 16);
  cvt(rgb_w1, R1t, 37, 16, 64, 16);
  cvt(rgb_w2, R2t, 16, 8, 32, 16);
  cvt(rgb_w3, R3t, 8, 1, 32, 16);

  k_prep<<<RS / 256, 256, 0, stream>>>(rgb_feat, neuray, ray_diff, mask,
                                       rd_w1, rd_b1, rd_w2, rd_b2,
                                       nr_w1, nr_b1, nr_w2, nr_b2,
                                       g_buf, feat_buf, w_buf, summask_buf);

  MlpArgs A;
  A.neuray = neuray; A.mask = mask; A.ray_diff = ray_diff;
  A.base_b1 = base_b1; A.base_b2 = base_b2;
  A.vis_b1 = vis_b1; A.vis_b2 = vis_b2;
  A.v2_b1 = v2_b1; A.v2_b2 = v2_b2;
  A.rgb_b1 = rgb_b1; A.rgb_b2 = rgb_b2; A.rgb_b3 = rgb_b3;
  A.W1t = W1t; A.W2t = W2t; A.VW1t = VW1t; A.VW2t = VW2t;
  A.U1t = U1t; A.U2t = U2t; A.R1t = R1t; A.R2t = R2t; A.R3t = R3t;
  A.g_buf = g_buf; A.feat_buf = feat_buf; A.w_buf = w_buf;
  A.x_buf = x_buf; A.visf_buf = visf_buf; A.blend_buf = blend_buf;
  // 524288 points / 16 per wave / 4 waves per block = 8192 blocks
  k_mlp<<<8192, 128, 0, stream>>>(A);

  k_geo<<<RS / 256, 256, 0, stream>>>(x_buf, visf_buf, geo_w1, geo_b1, geo_w2, geo_b2,
                                      pos_enc, gf_buf);

  k_attn<<<Rr, 64, 0, stream>>>(gf_buf, summask_buf, wq, wk, wv, att_fc,
                                ln_g, ln_b, og_w1, og_b1, og_w2, og_b2, sigma_buf);

  k_final<<<RS / 256, 256, 0, stream>>>(rgb_feat, mask, blend_buf, sigma_buf, (float*)d_out);
}
